// RNNModel_28578712387793
// MI455X (gfx1250) — compile-verified, matrix operand order checked by reading, not run on hardware
//
#include <hip/hip_runtime.h>
#include <hip/hip_bf16.h>

typedef __attribute__((ext_vector_type(16))) _Float16 v16h;
typedef __attribute__((ext_vector_type(8)))  float    v8f;

#define NTOKEN 10000
#define NINP   600
#define NHID   600
#define NB     6
#define TOPK   4
#define NOUT   10
#define BLK    100
#define KDIM   64
#define VDIM   100
#define CDIM   32
#define TSTEPS 256
#define BATCH  256
#define ROWS   16          // batch rows per workgroup (one WMMA M-tile)
#define NTHREADS 256
#define NWAVES 8

// padded dims (K multiples of 32, N multiples of 16)
#define NINP_P 608
#define BLK_P  128
#define VDIM_P 112
#define GATES  400
#define NOUT_P 16

// fragment geometry: one 16x16x32 operand tile = 32 lanes * 16 halves
#define FRAG   512
// k-tile counts
#define KT_INP 19          // 608/32
#define KT_BLK 4           // 128/32

// ---- f16 packed-weight workspace offsets (in halfs), fragment-order ----
// layout per weight: [batch][n_tile][k_tile][lane 32][elem 16]
#define OFF_WK   0
#define SZ_WK    (4*KT_INP*FRAG)            // Wk : NTN=4,  NTK=19
#define OFF_WV   (OFF_WK+SZ_WK)
#define SZ_WV    (7*KT_INP*FRAG)            // Wv : NTN=7,  NTK=19
#define OFF_WQ   (OFF_WV+SZ_WV)
#define SZ_WQ    (NB*4*KT_BLK*FRAG)         // Wq : NTN=4,  NTK=4, batch 6
#define OFF_WI   (OFF_WQ+SZ_WQ)
#define SZ_WI    (NB*25*KT_BLK*FRAG)        // Wi : NTN=25, NTK=4
#define OFF_WH   (OFF_WI+SZ_WI)
#define SZ_WH    (NB*25*KT_BLK*FRAG)        // Wh
#define OFF_WQC  (OFF_WH+SZ_WH)
#define SZ_WQC   (NB*2*KT_BLK*FRAG)         // Wqc: NTN=2,  NTK=4
#define OFF_WKC  (OFF_WQC+SZ_WQC)
#define SZ_WKC   (NB*2*KT_BLK*FRAG)
#define OFF_WVC  (OFF_WKC+SZ_WKC)
#define SZ_WVC   (NB*7*KT_BLK*FRAG)         // Wvc: NTN=7,  NTK=4
#define OFF_WD   (OFF_WVC+SZ_WVC)
#define SZ_WD    (1*KT_INP*FRAG)            // Wd : NTN=1,  NTK=19

// fragment index helpers: within a 32-wide k-tile,
//   lane = m(or n) + 16*hi,  hi = (kk>>3)&1,  elem = (kk&7) | ((kk&16)>>1)
// inverse: kk = (e&7) + 8*hi + ((e>>3)<<4)
__device__ __host__ __forceinline__ int frag_kk(int e, int hi) {
    return (e & 7) + (hi << 3) + ((e >> 3) << 4);
}

// fragment-order GEMM inner loop: A from LDS, B from global, both vector loads
__device__ __forceinline__ v8f wmma_frag(v8f acc, const _Float16* __restrict__ Afrag,
                                         const _Float16* __restrict__ Bfrag,
                                         int ktiles, int lane) {
    for (int kt = 0; kt < ktiles; ++kt) {
        v16h a = *(const v16h*)(Afrag + kt * FRAG + lane * 16);
        v16h b = *(const v16h*)(Bfrag + kt * FRAG + lane * 16);
        acc = __builtin_amdgcn_wmma_f32_16x16x32_f16(false, a, false, b, (short)0, acc,
                                                     false, false);
    }
    return acc;
}

// D/C 16x16 f32: VGPR r -> M = r + 8*(lane>=16), N = lane%16
__device__ __forceinline__ void store_d(float* D, int ldd, int lane, v8f acc) {
    int n  = lane & 15;
    int mb = (lane & 16) ? 8 : 0;
#pragma unroll
    for (int r = 0; r < 8; ++r) D[(r + mb) * ldd + n] = acc[r];
}

__device__ __forceinline__ float dsigmoid(float x) { return 1.f / (1.f + __expf(-x)); }

// ---------------- weight pack: fp32 [batch][K][N] -> fragment-order f16 ----------------
__global__ void pack_frag_kernel(const float* __restrict__ src, _Float16* __restrict__ dst,
                                 int K, int N, int NTK, int NTN, int batch) {
    long total = (long)batch * NTN * NTK * FRAG;
    for (long i = (long)blockIdx.x * blockDim.x + threadIdx.x; i < total;
         i += (long)gridDim.x * blockDim.x) {
        int  e    = (int)(i & 15);
        int  lane = (int)((i >> 4) & 31);
        long rt   = i >> 9;
        int  kt   = (int)(rt % NTK);
        long rn   = rt / NTK;
        int  nt   = (int)(rn % NTN);
        int  bt   = (int)(rn / NTN);
        int  nl = lane & 15, hi = lane >> 4;
        int  k = kt * 32 + frag_kk(e, hi);
        int  n = nt * 16 + nl;
        float v = (k < K && n < N) ? src[((long)bt * K + k) * N + n] : 0.f;
        dst[i] = (_Float16)v;
    }
}

// ---------------- main persistent scan kernel: one WG == 16 batch rows, all T steps ----
__global__ void __launch_bounds__(NTHREADS)
rim_scan_kernel(const int* __restrict__ input, const float* __restrict__ hx,
                const float* __restrict__ cx, const float* __restrict__ emb1,
                const float* __restrict__ emb2, const float* __restrict__ emb3,
                const float* __restrict__ bg, const float* __restrict__ bd,
                const _Float16* __restrict__ ws, float* __restrict__ out) {
    extern __shared__ char smem_raw[];
    float* hS    = (float*)smem_raw;             // [16][608] fp32 state h
    float* cS    = hS + ROWS * NINP_P;           // [16][608] fp32 state c
    float* outS  = cS + ROWS * NINP_P;           // [16][400] GEMM D staging
    float* vvS   = outS + ROWS * GATES;          // [16][112]
    float* kkS   = vvS + ROWS * VDIM_P;          // [16][64]
    float* probS = kkS + ROWS * KDIM;            // [16][8]
    float* maskS = probS + ROWS * 8;             // [16][8]
    float* qcS   = maskS + ROWS * 8;             // [16][6*32]
    float* kcS   = qcS + ROWS * NB * CDIM;       // [16][6*32]
    float* vcS   = kcS + ROWS * NB * CDIM;       // [16][6*112]
    float* attS  = vcS + ROWS * NB * VDIM_P;     // [16][6][8]
    // fragment-order f16 A staging (32B aligned: float prefix is 186368 B)
    _Float16* fragA = (_Float16*)(attS + ROWS * NB * 8);   // max(19, 6*4) k-tiles = 12288 halfs
    _Float16* inpA  = fragA + NB * KT_BLK * FRAG;          // 4 k-tiles = 2048 halfs

    const int tid  = threadIdx.x;
    const int lane = tid & 31;
    const int wave = tid >> 5;
    const int r0   = blockIdx.x * ROWS;

    const _Float16* WkP  = ws + OFF_WK;
    const _Float16* WvP  = ws + OFF_WV;
    const _Float16* WqP  = ws + OFF_WQ;
    const _Float16* WiP  = ws + OFF_WI;
    const _Float16* WhP  = ws + OFF_WH;
    const _Float16* WqcP = ws + OFF_WQC;
    const _Float16* WkcP = ws + OFF_WKC;
    const _Float16* WvcP = ws + OFF_WVC;
    const _Float16* WdP  = ws + OFF_WD;

    // ---- init state from hx/cx ----
    for (int i = tid; i < ROWS * NINP_P; i += NTHREADS) {
        int b = i / NINP_P, j = i - b * NINP_P;
        float hv = 0.f, cv = 0.f;
        if (j < NHID) {
            hv = hx[(r0 + b) * NHID + j];
            cv = cx[(r0 + b) * NHID + j];
        }
        hS[i] = hv;
        cS[i] = cv;
    }
    __syncthreads();

    for (int t = 0; t < TSTEPS; ++t) {
        // 1) embedding gather -> fragA (fragment order, K padded 600->608)
        const int* ip = input + ((size_t)t * BATCH + r0) * 3;
        for (int i = tid; i < KT_INP * FRAG; i += NTHREADS) {
            int e = i & 15, ln = (i >> 4) & 31, kt = i >> 9;
            int m = ln & 15, hi = ln >> 4;
            int k = kt * 32 + frag_kk(e, hi);
            float v = 0.f;
            if (k < NINP) {
                int seg = k / 200, off = k - seg * 200;
                int tok = ip[m * 3 + seg];
                const float* tab = (seg == 0) ? emb1 : (seg == 1) ? emb2 : emb3;
                v = tab[tok * 200 + off];
            }
            fragA[i] = (_Float16)v;
        }
        __syncthreads();

        // 2) kk = emb@Wk (4 tiles), vv = emb@Wv (7 tiles)
        for (int w = wave; w < 11; w += NWAVES) {
            v8f acc = {};
            if (w < 4) {
                acc = wmma_frag(acc, fragA, WkP + w * (KT_INP * FRAG), KT_INP, lane);
                store_d(kkS + w * 16, KDIM, lane, acc);
            } else {
                int tt = w - 4;
                acc = wmma_frag(acc, fragA, WvP + tt * (KT_INP * FRAG), KT_INP, lane);
                store_d(vvS + tt * 16, VDIM_P, lane, acc);
            }
        }
        __syncthreads();

        // 3) stage per-block h (old) as fragment-order A, K padded 100->128
        for (int i = tid; i < NB * KT_BLK * FRAG; i += NTHREADS) {
            int e = i & 15, ln = (i >> 4) & 31;
            int kt = (i >> 9) & 3, n = i >> 11;
            int m = ln & 15, hi = ln >> 4;
            int k = kt * 32 + frag_kk(e, hi);
            fragA[i] = (_Float16)((k < BLK) ? hS[m * NINP_P + n * BLK + k] : 0.f);
        }
        __syncthreads();

        // 4) q = hb@Wq : 6 blocks x 4 tiles
        for (int w = wave; w < 24; w += NWAVES) {
            int n = w >> 2, tt = w & 3;
            v8f acc = {};
            acc = wmma_frag(acc, fragA + n * (KT_BLK * FRAG),
                            WqP + (n * 4 + tt) * (KT_BLK * FRAG), KT_BLK, lane);
            store_d(outS + n * KDIM + tt * 16, NB * KDIM, lane, acc);
        }
        __syncthreads();

        // 5a) probs0 = softmax([q.kk/sqrt(K), 0])[0] = sigmoid(q.kk/8)
        if (tid < ROWS * NB) {
            int b = tid / NB, n = tid - b * NB;
            float s = 0.f;
            for (int d = 0; d < KDIM; ++d)
                s += outS[b * (NB * KDIM) + n * KDIM + d] * kkS[b * KDIM + d];
            probS[b * 8 + n] = dsigmoid(s * 0.125f);
        }
        __syncthreads();
        // 5b) top-4 mask (rank with index tie-break, matches jax top_k)
        if (tid < ROWS) {
            int b = tid;
            for (int n = 0; n < NB; ++n) {
                int rank = 0;
                float pn = probS[b * 8 + n];
                for (int m = 0; m < NB; ++m) {
                    float pm = probS[b * 8 + m];
                    if (pm > pn || (pm == pn && m < n)) rank++;
                }
                maskS[b * 8 + n] = (rank < TOPK) ? 1.f : 0.f;
            }
        }
        __syncthreads();

        // 6) per-block LSTM: gates = inp_att@Wi + hb@Wh + bg
        for (int n = 0; n < NB; ++n) {
            for (int i = tid; i < KT_BLK * FRAG; i += NTHREADS) {
                int e = i & 15, ln = (i >> 4) & 31, kt = i >> 9;
                int m = ln & 15, hi = ln >> 4;
                int k = kt * 32 + frag_kk(e, hi);
                inpA[i] = (_Float16)((k < VDIM) ? probS[m * 8 + n] * vvS[m * VDIM_P + k]
                                                : 0.f);
            }
            __syncthreads();
            for (int w = wave; w < 25; w += NWAVES) {  // 400/16 N tiles
                v8f acc = {};
                acc = wmma_frag(acc, inpA, WiP + (n * 25 + w) * (KT_BLK * FRAG),
                                KT_BLK, lane);
                acc = wmma_frag(acc, fragA + n * (KT_BLK * FRAG),
                                WhP + (n * 25 + w) * (KT_BLK * FRAG), KT_BLK, lane);
                store_d(outS + w * 16, GATES, lane, acc);
            }
            __syncthreads();
            for (int i = tid; i < ROWS * BLK; i += NTHREADS) {
                int b = i / BLK, j = i - b * BLK;
                const float* gb  = outS + b * GATES;
                const float* bgn = bg + n * GATES;
                float si = dsigmoid(gb[j] + bgn[j]);
                float sf = dsigmoid(gb[BLK + j] + bgn[BLK + j]);
                float tg = tanhf(gb[2 * BLK + j] + bgn[2 * BLK + j]);
                float so = dsigmoid(gb[3 * BLK + j] + bgn[3 * BLK + j]);
                float co = cS[b * NINP_P + n * BLK + j];
                float cn = sf * co + si * tg;
                float hn = so * tanhf(cn);
                float m  = maskS[b * 8 + n];
                float ho = hS[b * NINP_P + n * BLK + j];
                cS[b * NINP_P + n * BLK + j] = m * cn + (1.f - m) * co;
                hS[b * NINP_P + n * BLK + j] = m * hn + (1.f - m) * ho;
            }
            __syncthreads();
        }

        // 7) restage h_upd blocks as fragment-order A
        for (int i = tid; i < NB * KT_BLK * FRAG; i += NTHREADS) {
            int e = i & 15, ln = (i >> 4) & 31;
            int kt = (i >> 9) & 3, n = i >> 11;
            int m = ln & 15, hi = ln >> 4;
            int k = kt * 32 + frag_kk(e, hi);
            fragA[i] = (_Float16)((k < BLK) ? hS[m * NINP_P + n * BLK + k] : 0.f);
        }
        __syncthreads();

        // 8) qc (12 tiles), kc (12), vc (42)
        for (int w = wave; w < 66; w += NWAVES) {
            v8f acc = {};
            if (w < 12) {
                int n = w >> 1, tt = w & 1;
                acc = wmma_frag(acc, fragA + n * (KT_BLK * FRAG),
                                WqcP + (n * 2 + tt) * (KT_BLK * FRAG), KT_BLK, lane);
                store_d(qcS + n * CDIM + tt * 16, NB * CDIM, lane, acc);
            } else if (w < 24) {
                int u = w - 12, n = u >> 1, tt = u & 1;
                acc = wmma_frag(acc, fragA + n * (KT_BLK * FRAG),
                                WkcP + (n * 2 + tt) * (KT_BLK * FRAG), KT_BLK, lane);
                store_d(kcS + n * CDIM + tt * 16, NB * CDIM, lane, acc);
            } else {
                int u = w - 24, n = u / 7, tt = u - n * 7;
                acc = wmma_frag(acc, fragA + n * (KT_BLK * FRAG),
                                WvcP + (n * 7 + tt) * (KT_BLK * FRAG), KT_BLK, lane);
                store_d(vcS + n * VDIM_P + tt * 16, NB * VDIM_P, lane, acc);
            }
        }
        __syncthreads();

        // 9) communication attention softmax (6x6, VALU)
        if (tid < ROWS * NB) {
            int b = tid / NB, n = tid - b * NB;
            float e[NB];
            float mx = -1e30f;
            for (int m = 0; m < NB; ++m) {
                float s = 0.f;
                for (int d = 0; d < CDIM; ++d)
                    s += qcS[b * (NB * CDIM) + n * CDIM + d] *
                         kcS[b * (NB * CDIM) + m * CDIM + d];
                e[m] = s * 0.17677669529663689f;  // 1/sqrt(32)
                mx = fmaxf(mx, e[m]);
            }
            float sum = 0.f;
            for (int m = 0; m < NB; ++m) { e[m] = __expf(e[m] - mx); sum += e[m]; }
            float inv = 1.f / sum;
            for (int m = 0; m < NB; ++m) attS[(b * NB + n) * 8 + m] = e[m] * inv;
        }
        __syncthreads();

        // 10) h += mask * (att @ vc)
        for (int i = tid; i < ROWS * NHID; i += NTHREADS) {
            int b = i / NHID, r = i - b * NHID;
            int n = r / BLK, j = r - n * BLK;
            float acc = 0.f;
            for (int m = 0; m < NB; ++m)
                acc += attS[(b * NB + n) * 8 + m] * vcS[b * (NB * VDIM_P) + m * VDIM_P + j];
            hS[b * NINP_P + r] += maskS[b * 8 + n] * acc;
        }
        __syncthreads();

        // 11) decoder: dec = h_fin @ Wd + bd (full 600-wide rows, fragment order)
        for (int i = tid; i < KT_INP * FRAG; i += NTHREADS) {
            int e = i & 15, ln = (i >> 4) & 31, kt = i >> 9;
            int m = ln & 15, hi = ln >> 4;
            int k = kt * 32 + frag_kk(e, hi);
            fragA[i] = (_Float16)((k < NHID) ? hS[m * NINP_P + k] : 0.f);
        }
        __syncthreads();
        if (wave == 0) {
            v8f acc = {};
            acc = wmma_frag(acc, fragA, WdP, KT_INP, lane);
            store_d(outS, NOUT_P, lane, acc);
        }
        __syncthreads();
        if (tid < ROWS * NOUT) {
            int b = tid / NOUT, j = tid - b * NOUT;
            out[(size_t)t * (BATCH * NOUT) + (size_t)(r0 + b) * NOUT + j] =
                outS[b * NOUT_P + j] + bd[j];
        }
        __syncthreads();
    }

    // final hx, cx
    const size_t DEC = (size_t)TSTEPS * BATCH * NOUT;
    for (int i = tid; i < ROWS * NHID; i += NTHREADS) {
        int b = i / NHID, j = i - b * NHID;
        out[DEC + (size_t)(r0 + b) * NHID + j] = hS[b * NINP_P + j];
        out[DEC + (size_t)BATCH * NHID + (size_t)(r0 + b) * NHID + j] = cS[b * NINP_P + j];
    }
}

static void launch_pack(const float* s, _Float16* dst, int K, int N, int NTK, int NTN,
                        int batch, hipStream_t stream) {
    long total = (long)batch * NTN * NTK * FRAG;
    int  grid  = (int)((total + 255) / 256);
    if (grid > 1024) grid = 1024;
    pack_frag_kernel<<<grid, 256, 0, stream>>>(s, dst, K, N, NTK, NTN, batch);
}

extern "C" void kernel_launch(void* const* d_in, const int* in_sizes, int n_in,
                              void* d_out, int out_size, void* d_ws, size_t ws_size,
                              hipStream_t stream) {
    (void)in_sizes; (void)n_in; (void)out_size; (void)ws_size;
    const int*   input = (const int*)d_in[0];
    const float* hx    = (const float*)d_in[1];
    const float* cx    = (const float*)d_in[2];
    const float* emb1  = (const float*)d_in[3];
    const float* emb2  = (const float*)d_in[4];
    const float* emb3  = (const float*)d_in[5];
    const float* Wk    = (const float*)d_in[6];
    const float* Wv    = (const float*)d_in[7];
    const float* Wq    = (const float*)d_in[8];
    const float* Wi    = (const float*)d_in[9];
    const float* Wh    = (const float*)d_in[10];
    const float* bg    = (const float*)d_in[11];
    const float* Wqc   = (const float*)d_in[12];
    const float* Wkc   = (const float*)d_in[13];
    const float* Wvc   = (const float*)d_in[14];
    const float* Wd    = (const float*)d_in[15];
    const float* bd    = (const float*)d_in[16];
    _Float16*    ws    = (_Float16*)d_ws;

    launch_pack(Wk,  ws + OFF_WK,  NINP, KDIM,  KT_INP, 4,  1,  stream);
    launch_pack(Wv,  ws + OFF_WV,  NINP, VDIM,  KT_INP, 7,  1,  stream);
    launch_pack(Wq,  ws + OFF_WQ,  BLK,  KDIM,  KT_BLK, 4,  NB, stream);
    launch_pack(Wi,  ws + OFF_WI,  VDIM, GATES, KT_BLK, 25, NB, stream);
    launch_pack(Wh,  ws + OFF_WH,  BLK,  GATES, KT_BLK, 25, NB, stream);
    launch_pack(Wqc, ws + OFF_WQC, BLK,  CDIM,  KT_BLK, 2,  NB, stream);
    launch_pack(Wkc, ws + OFF_WKC, BLK,  CDIM,  KT_BLK, 2,  NB, stream);
    launch_pack(Wvc, ws + OFF_WVC, BLK,  BLK,   KT_BLK, 7,  NB, stream);
    launch_pack(Wd,  ws + OFF_WD,  NINP, NOUT,  KT_INP, 1,  1,  stream);

    size_t lds = (size_t)ROWS *
                     (2 * NINP_P + GATES + VDIM_P + KDIM + 16 + 2 * NB * CDIM +
                      NB * VDIM_P + NB * 8) * sizeof(float) +
                 (size_t)(NB * KT_BLK * FRAG + KT_BLK * FRAG) * sizeof(_Float16);

    rim_scan_kernel<<<dim3(BATCH / ROWS), dim3(NTHREADS), lds, stream>>>(
        input, hx, cx, emb1, emb2, emb3, bg, bd, ws, (float*)d_out);
}
